// TinyRecursiveModelTRMv6_2207613190483
// MI455X (gfx1250) — matrix operand order repaired and checked by reading, not compiled, not run on hardware
//
#include <hip/hip_runtime.h>
#include <math.h>

// ---------------------------------------------------------------------------
// Types for CDNA5 WMMA (gfx1250, wave32)
// ---------------------------------------------------------------------------
typedef __attribute__((ext_vector_type(16))) __bf16 v16bf;
typedef __attribute__((ext_vector_type(8)))  float  v8f;

constexpr int EPI_F32       = 0;  // out f32 = acc + bias
constexpr int EPI_BF16      = 1;  // out bf16 = acc + bias
constexpr int EPI_GELU_BF16 = 2;  // out bf16 = gelu(acc + bias), exact erf GELU
constexpr int EPI_TANH_F32  = 3;  // out f32 = tanh(acc + bias)

// ---------------------------------------------------------------------------
// Tiled bf16 GEMM, v_wmma_f32_16x16x32_bf16, async-LDS double buffered.
//
//   C[M,N] = A[M,K] @ W[K,N] + bias,  W passed PRE-TRANSPOSED as Wt[N,K].
//
// Block tile 128(M) x 128(N), 256 threads = 8 waves; wave tile 32(M) x 64(N)
// = 2x4 WMMA 16x16 tiles, K stepped by 32 -> 8 WMMAs per wave per K-step.
//
// Staging: global -> LDS via GLOBAL_LOAD_ASYNC_TO_LDS_B128 (ASYNCcnt), two
// LDS buffers (2 x (A 128x40 + B 128x40) bf16 = 40960 B dynamic LDS). Next
// tile's async copies are issued before computing the current tile; one
// s_wait_asynccnt + one barrier per K-step.
//
// Fragment layouts follow CDNA5 ISA 16-bit A (16x32) / B (32x16):
//   A: lane<16 -> K in [0,8)U[16,24), lane>=16 -> [8,16)U[24,32); M=lane%16
//   B: lane<16 -> K in [0,8)U[16,24), lane>=16 -> [8,16)U[24,32); N=lane%16
// Row stride 40 elems (80 B) keeps every fragment b128-aligned.
// ---------------------------------------------------------------------------
__device__ __forceinline__ void async_ld_b128(unsigned lds_off, const void* gaddr) {
  asm volatile("global_load_async_to_lds_b128 %0, %1, off"
               :: "v"(lds_off), "v"(gaddr) : "memory");
}
__device__ __forceinline__ void wait_async0() {
  asm volatile("s_wait_asynccnt 0x0" ::: "memory");
}

template <int EPI>
__global__ __launch_bounds__(256) void gemm_bf16_wmma(
    const __bf16* __restrict__ A, const __bf16* __restrict__ Wt,
    const float* __restrict__ bias, void* __restrict__ outv,
    int M, int N, int Kd, int ldOut)
{
  extern __shared__ char smem[];  // 40960 bytes, dynamic => LDS offset base 0

  constexpr int LDT  = 40;              // bf16 elems per LDS row (80 B, 16B-aligned)
  constexpr int BUFB = 128 * LDT * 2;   // 10240 bytes per matrix buffer

  const int tid  = threadIdx.x;
  const int lane = tid & 31;
  const int wave = tid >> 5;
  const int wm   = wave >> 1;           // 0..3 (M)
  const int wn   = wave & 1;            // 0..1 (N)
  const int m0   = blockIdx.x * 128;
  const int n0   = blockIdx.y * 128;

  // byte offsets inside dynamic LDS
  auto ofsA = [&](int buf, int row, int k) -> int { return buf * BUFB + (row * LDT + k) * 2; };
  auto ofsB = [&](int buf, int n,   int k) -> int { return 2 * BUFB + buf * BUFB + (n * LDT + k) * 2; };

  const int srow = tid >> 1;            // staging row 0..127 (A: m, B: n)
  const int scol = (tid & 1) * 16;      // 0 or 16

  auto issue_tile = [&](int k0, int buf) {
    const __bf16* ga = A  + (size_t)(m0 + srow) * Kd + k0 + scol;
    const __bf16* gb = Wt + (size_t)(n0 + srow) * Kd + k0 + scol;
    const unsigned la = (unsigned)ofsA(buf, srow, scol);
    const unsigned lb = (unsigned)ofsB(buf, srow, scol);
    async_ld_b128(la,       ga);
    async_ld_b128(la + 16u, ga + 8);
    async_ld_b128(lb,       gb);
    async_ld_b128(lb + 16u, gb + 8);
  };

  v8f acc[2][4];
#pragma unroll
  for (int i = 0; i < 2; ++i)
#pragma unroll
    for (int j = 0; j < 4; ++j)
#pragma unroll
      for (int v = 0; v < 8; ++v) acc[i][j][v] = 0.0f;

  // prologue: tile 0 into buffer 0
  issue_tile(0, 0);
  wait_async0();
  __syncthreads();

  int p = 0;
  for (int k0 = 0; k0 < Kd; k0 += 32) {
    const bool hasNext = (k0 + 32) < Kd;
    if (hasNext) issue_tile(k0 + 32, p ^ 1);  // safe: all waves done reading p^1

    union { v16bf v; uint4 u[2]; } afr[2], bfr[4];
    const int kba = (lane < 16) ? 0 : 8;
#pragma unroll
    for (int mt = 0; mt < 2; ++mt) {
      const int row = wm * 32 + mt * 16 + (lane & 15);
      afr[mt].u[0] = *reinterpret_cast<const uint4*>(smem + ofsA(p, row, kba));
      afr[mt].u[1] = *reinterpret_cast<const uint4*>(smem + ofsA(p, row, kba + 16));
    }
    const int kbb = (lane < 16) ? 0 : 8;
#pragma unroll
    for (int nt = 0; nt < 4; ++nt) {
      const int col = wn * 64 + nt * 16 + (lane & 15);
      bfr[nt].u[0] = *reinterpret_cast<const uint4*>(smem + ofsB(p, col, kbb));
      bfr[nt].u[1] = *reinterpret_cast<const uint4*>(smem + ofsB(p, col, kbb + 16));
    }

#pragma unroll
    for (int mt = 0; mt < 2; ++mt)
#pragma unroll
      for (int nt = 0; nt < 4; ++nt)
        acc[mt][nt] = __builtin_amdgcn_wmma_f32_16x16x32_bf16(
            false, afr[mt].v, false, bfr[nt].v, (short)0, acc[mt][nt],
            false, false);

    if (hasNext) wait_async0();
    __syncthreads();
    p ^= 1;
  }

  // epilogue: C/D layout, lane<16 -> M=v, lane>=16 -> M=v+8, N=lane%16
#pragma unroll
  for (int mt = 0; mt < 2; ++mt) {
#pragma unroll
    for (int nt = 0; nt < 4; ++nt) {
      const int gn     = n0 + wn * 64 + nt * 16 + (lane & 15);
      const int gmBase = m0 + wm * 32 + mt * 16 + ((lane < 16) ? 0 : 8);
      const float bv   = bias ? bias[gn] : 0.0f;
#pragma unroll
      for (int v = 0; v < 8; ++v) {
        const int gm = gmBase + v;
        float val = acc[mt][nt][v] + bv;
        if (EPI == EPI_GELU_BF16)
          val = 0.5f * val * (1.0f + erff(val * 0.70710678118654752f));
        if (EPI == EPI_TANH_F32) val = tanhf(val);
        const size_t oidx = (size_t)gm * ldOut + gn;
        if (EPI == EPI_F32 || EPI == EPI_TANH_F32)
          ((float*)outv)[oidx] = val;
        else
          ((__bf16*)outv)[oidx] = (__bf16)val;
      }
    }
  }
}

// ---------------------------------------------------------------------------
// Block-per-row reduction helper (256 threads)
// ---------------------------------------------------------------------------
__device__ __forceinline__ float blockReduceSum(float v, float* sh) {
  const int t = threadIdx.x;
  sh[t] = v;
  __syncthreads();
#pragma unroll
  for (int s = 128; s > 0; s >>= 1) {
    if (t < s) sh[t] += sh[t + s];
    __syncthreads();
  }
  const float r = sh[0];
  __syncthreads();
  return r;
}

// LN(concat of NS 1024-wide f32 sources) -> bf16 out, one block per row.
template <int NS>
__global__ __launch_bounds__(256) void catln_kernel(
    const float* __restrict__ a0, const float* __restrict__ a1,
    const float* __restrict__ a2, const float* __restrict__ g,
    const float* __restrict__ b, __bf16* __restrict__ out)
{
  const int W = NS * 1024;
  const int r = blockIdx.x, tid = threadIdx.x;
  __shared__ float sh[256];
  float s = 0.f, ss = 0.f;
  for (int i = tid; i < W; i += 256) {
    const int seg = i >> 10, off = i & 1023;
    const float* src = (seg == 0) ? a0 : ((seg == 1) ? a1 : a2);
    const float v = src[(size_t)r * 1024 + off];
    s += v; ss += v * v;
  }
  const float S1 = blockReduceSum(s, sh);
  const float S2 = blockReduceSum(ss, sh);
  const float mu = S1 / (float)W;
  const float var = S2 / (float)W - mu * mu;
  const float rs = rsqrtf(var + 1e-5f);
  for (int i = tid; i < W; i += 256) {
    const int seg = i >> 10, off = i & 1023;
    const float* src = (seg == 0) ? a0 : ((seg == 1) ? a1 : a2);
    const float v = src[(size_t)r * 1024 + off];
    out[(size_t)r * W + i] = (__bf16)((v - mu) * rs * g[i] + b[i]);
  }
}

// z = LN(z + scale * t) with ln_z gains; one block per row.
__global__ __launch_bounds__(256) void zresln_kernel(
    float* __restrict__ z, const float* __restrict__ t, float scale,
    const float* __restrict__ g, const float* __restrict__ b)
{
  const int r = blockIdx.x, tid = threadIdx.x;
  __shared__ float sh[256];
  float s = 0.f, ss = 0.f;
  for (int j = tid; j < 1024; j += 256) {
    const size_t idx = (size_t)r * 1024 + j;
    const float v = z[idx] + scale * t[idx];
    s += v; ss += v * v;
  }
  const float S1 = blockReduceSum(s, sh);
  const float S2 = blockReduceSum(ss, sh);
  const float mu = S1 / 1024.f;
  const float var = S2 / 1024.f - mu * mu;
  const float rs = rsqrtf(var + 1e-5f);
  for (int j = tid; j < 1024; j += 256) {
    const size_t idx = (size_t)r * 1024 + j;
    const float v = z[idx] + scale * t[idx];
    z[idx] = (v - mu) * rs * g[j] + b[j];
  }
}

// al = tanh(a @ aff_W + aff_b), a:[B,3], aff_W:[3,1024]
__global__ void affect_kernel(const float* __restrict__ a,
                              const float* __restrict__ W,
                              const float* __restrict__ b,
                              float* __restrict__ al, size_t n)
{
  const size_t i = (size_t)blockIdx.x * 256 + threadIdx.x;
  if (i >= n) return;
  const size_t r = i >> 10;
  const int d = (int)(i & 1023);
  const float v = a[r * 3 + 0] * W[d] + a[r * 3 + 1] * W[1024 + d] +
                  a[r * 3 + 2] * W[2048 + d] + b[d];
  al[i] = tanhf(v);
}

// g = sigmoid([z,al] @ gate_W + gate_b); z += 0.3*g*al; write bf16 mirror; conf += g
__global__ __launch_bounds__(256) void gate_update_kernel(
    float* __restrict__ z, const float* __restrict__ al,
    const float* __restrict__ gw, const float* __restrict__ gb,
    __bf16* __restrict__ zbf, float* __restrict__ confslot)
{
  const int r = blockIdx.x, tid = threadIdx.x;
  __shared__ float sh[256];
  float p = 0.f;
  for (int j = tid; j < 1024; j += 256)
    p += z[(size_t)r * 1024 + j] * gw[j] + al[(size_t)r * 1024 + j] * gw[1024 + j];
  const float dot = blockReduceSum(p, sh);
  const float g = 1.f / (1.f + expf(-(dot + gb[0])));
  for (int j = tid; j < 1024; j += 256) {
    const size_t idx = (size_t)r * 1024 + j;
    const float nz = z[idx] + 0.3f * g * al[idx];
    z[idx] = nz;
    zbf[idx] = (__bf16)nz;
  }
  if (tid == 0) atomicAdd(confslot, g);
}

// softmax over S cached K rows, combine V rows -> o (bf16). One block per row.
__global__ __launch_bounds__(256) void attn_combine_kernel(
    const __bf16* __restrict__ q, const __bf16* __restrict__ Kc,
    const __bf16* __restrict__ Vc, int S, __bf16* __restrict__ obf)
{
  const int r = blockIdx.x, tid = threadIdx.x;
  __shared__ float sh[256];
  __shared__ float att[8];
  for (int s = 0; s < S; ++s) {
    const __bf16* kp = Kc + ((size_t)r * 5 + s) * 1024;
    const __bf16* qp = q + (size_t)r * 1024;
    float p = 0.f;
    for (int j = tid; j < 1024; j += 256) p += (float)qp[j] * (float)kp[j];
    const float d = blockReduceSum(p, sh);
    if (tid == 0) att[s] = d * (1.0f / 32.0f);  // 1/sqrt(1024)
    __syncthreads();
  }
  if (tid == 0) {
    float mx = att[0];
    for (int s = 1; s < S; ++s) mx = fmaxf(mx, att[s]);
    float se = 0.f;
    for (int s = 0; s < S; ++s) { att[s] = expf(att[s] - mx); se += att[s]; }
    for (int s = 0; s < S; ++s) att[s] /= se;
  }
  __syncthreads();
  for (int j = tid; j < 1024; j += 256) {
    float o = 0.f;
    for (int s = 0; s < S; ++s)
      o += att[s] * (float)Vc[((size_t)r * 5 + s) * 1024 + j];
    obf[(size_t)r * 1024 + j] = (__bf16)o;
  }
}

// cat4 = bf16([x, y, z, tanh(z+y)])
__global__ void soulcat_kernel(const float* __restrict__ x,
                               const float* __restrict__ y,
                               const float* __restrict__ z,
                               __bf16* __restrict__ cat, size_t n)
{
  const size_t i = (size_t)blockIdx.x * 256 + threadIdx.x;
  if (i >= n) return;
  const size_t r = i >> 12;
  const int rem = (int)(i & 4095);
  const int seg = rem >> 10, off = rem & 1023;
  const size_t idx = r * 1024 + off;
  float v;
  if (seg == 0)      v = x[idx];
  else if (seg == 1) v = y[idx];
  else if (seg == 2) v = z[idx];
  else               v = tanhf(z[idx] + y[idx]);
  cat[i] = (__bf16)v;
}

// gs = sigmoid(m @ soul_gW + gb); z += 0.3*gs*m
__global__ __launch_bounds__(256) void soulgate_kernel(
    float* __restrict__ z, const float* __restrict__ m,
    const float* __restrict__ gw, const float* __restrict__ gb)
{
  const int r = blockIdx.x, tid = threadIdx.x;
  __shared__ float sh[256];
  float p = 0.f;
  for (int j = tid; j < 1024; j += 256) p += m[(size_t)r * 1024 + j] * gw[j];
  const float dot = blockReduceSum(p, sh);
  const float gs = 1.f / (1.f + expf(-(dot + gb[0])));
  for (int j = tid; j < 1024; j += 256) {
    const size_t idx = (size_t)r * 1024 + j;
    z[idx] += 0.3f * gs * m[idx];
  }
}

// pad = tanh(z @ pad_W + pad_b), pad_W:[1024,3]
__global__ __launch_bounds__(256) void pad_kernel(
    const float* __restrict__ z, const float* __restrict__ pw,
    const float* __restrict__ pb, float* __restrict__ out)
{
  const int r = blockIdx.x, tid = threadIdx.x;
  __shared__ float sh[256];
  float p0 = 0.f, p1 = 0.f, p2 = 0.f;
  for (int j = tid; j < 1024; j += 256) {
    const float zv = z[(size_t)r * 1024 + j];
    p0 += zv * pw[j * 3 + 0];
    p1 += zv * pw[j * 3 + 1];
    p2 += zv * pw[j * 3 + 2];
  }
  const float d0 = blockReduceSum(p0, sh);
  const float d1 = blockReduceSum(p1, sh);
  const float d2 = blockReduceSum(p2, sh);
  if (tid == 0) {
    out[(size_t)r * 3 + 0] = tanhf(d0 + pb[0]);
    out[(size_t)r * 3 + 1] = tanhf(d1 + pb[1]);
    out[(size_t)r * 3 + 2] = tanhf(d2 + pb[2]);
  }
}

// weight convert + transpose: src[K,N] f32 -> dst[N,K] bf16 (done once per call;
// bf16 weights then stay L2-resident: 46 MB << 192 MB L2)
__global__ void wt_kernel(const float* __restrict__ s, __bf16* __restrict__ d,
                          int K, int N) {
  const size_t i = (size_t)blockIdx.x * 256 + threadIdx.x;
  if (i >= (size_t)K * N) return;
  const int k = (int)(i / N), n = (int)(i % N);
  d[(size_t)n * K + k] = (__bf16)s[i];
}
__global__ void copy_kernel(const float* __restrict__ s, float* __restrict__ d, size_t n) {
  const size_t i = (size_t)blockIdx.x * 256 + threadIdx.x;
  if (i < n) d[i] = s[i];
}
__global__ void zero_kernel(float* __restrict__ d, size_t n) {
  const size_t i = (size_t)blockIdx.x * 256 + threadIdx.x;
  if (i < n) d[i] = 0.f;
}
__global__ void axpy_kernel(float* __restrict__ y, const float* __restrict__ t,
                            float scale, size_t n) {
  const size_t i = (size_t)blockIdx.x * 256 + threadIdx.x;
  if (i < n) y[i] += scale * t[i];
}
__global__ void conf_final_kernel(const float* __restrict__ c, float* __restrict__ o) {
  const int t = threadIdx.x;
  if (t < 5) o[t] = c[t] * (1.0f / 8192.0f);
}

// ---------------------------------------------------------------------------
// Host orchestration
// ---------------------------------------------------------------------------
static void launch_gemm(int epi, const __bf16* A, const __bf16* Wt,
                        const float* bias, void* out, int M, int N, int Kd,
                        int ldOut, hipStream_t s)
{
  dim3 g(M / 128, N / 128), b(256);
  const size_t shmem = 40960;
  switch (epi) {
    case EPI_F32:
      gemm_bf16_wmma<EPI_F32><<<g, b, shmem, s>>>(A, Wt, bias, out, M, N, Kd, ldOut); break;
    case EPI_BF16:
      gemm_bf16_wmma<EPI_BF16><<<g, b, shmem, s>>>(A, Wt, bias, out, M, N, Kd, ldOut); break;
    case EPI_GELU_BF16:
      gemm_bf16_wmma<EPI_GELU_BF16><<<g, b, shmem, s>>>(A, Wt, bias, out, M, N, Kd, ldOut); break;
    default:
      gemm_bf16_wmma<EPI_TANH_F32><<<g, b, shmem, s>>>(A, Wt, bias, out, M, N, Kd, ldOut); break;
  }
}

extern "C" void kernel_launch(void* const* d_in, const int* in_sizes, int n_in,
                              void* d_out, int out_size, void* d_ws, size_t ws_size,
                              hipStream_t stream)
{
  (void)in_sizes; (void)n_in; (void)out_size; (void)ws_size;
  constexpr int B = 8192, D = 1024, KIT = 5;

  const float* x    = (const float*)d_in[0];
  const float* y0   = (const float*)d_in[1];
  const float* aIn  = (const float*)d_in[2];
  // d_in[3] = K (static 5)
  const float* ln_lat_g = (const float*)d_in[4];
  const float* ln_lat_b = (const float*)d_in[5];
  const float* ln_ans_g = (const float*)d_in[6];
  const float* ln_ans_b = (const float*)d_in[7];
  const float* ln_z_g   = (const float*)d_in[8];
  const float* ln_z_b   = (const float*)d_in[9];
  const float* lat_W1 = (const float*)d_in[10]; const float* lat_b1 = (const float*)d_in[11];
  const float* lat_W2 = (const float*)d_in[12]; const float* lat_b2 = (const float*)d_in[13];
  const float* ans_W1 = (const float*)d_in[14]; const float* ans_b1 = (const float*)d_in[15];
  const float* ans_W2 = (const float*)d_in[16]; const float* ans_b2 = (const float*)d_in[17];
  const float* aff_W  = (const float*)d_in[18]; const float* aff_b  = (const float*)d_in[19];
  const float* gate_W = (const float*)d_in[20]; const float* gate_b = (const float*)d_in[21];
  const float* Wq = (const float*)d_in[22]; const float* bq = (const float*)d_in[23];
  const float* Wk = (const float*)d_in[24]; const float* bk = (const float*)d_in[25];
  const float* Wv = (const float*)d_in[26]; const float* bvb = (const float*)d_in[27];
  const float* Wo = (const float*)d_in[28]; const float* bo = (const float*)d_in[29];
  const float* pad_W = (const float*)d_in[30]; const float* pad_b = (const float*)d_in[31];
  const float* soul_W = (const float*)d_in[32]; const float* soul_b = (const float*)d_in[33];
  const float* soul_gW = (const float*)d_in[34]; const float* soul_gb = (const float*)d_in[35];

  float* outY    = (float*)d_out;
  float* outConf = outY + (size_t)B * D;
  float* outPad  = outConf + KIT;

  // workspace carve-out (256B aligned)
  char* ws = (char*)d_ws;
  size_t off = 0;
  auto alloc = [&](size_t bytes) -> void* {
    void* p = ws + off;
    off += (bytes + 255) & ~(size_t)255;
    return p;
  };
  float*  z    = (float*)alloc((size_t)B * D * 4);
  float*  al   = (float*)alloc((size_t)B * D * 4);
  float*  tmp  = (float*)alloc((size_t)B * D * 4);
  __bf16* cat  = (__bf16*)alloc((size_t)B * 4 * D * 2);
  __bf16* h1   = (__bf16*)alloc((size_t)B * 2 * D * 2);
  __bf16* zbf  = (__bf16*)alloc((size_t)B * D * 2);
  __bf16* qbf  = (__bf16*)alloc((size_t)B * D * 2);
  __bf16* obf  = (__bf16*)alloc((size_t)B * D * 2);
  __bf16* Kc   = (__bf16*)alloc((size_t)B * 5 * D * 2);
  __bf16* Vc   = (__bf16*)alloc((size_t)B * 5 * D * 2);
  __bf16* wlat1 = (__bf16*)alloc((size_t)3 * D * 2 * D * 2);   // [2D x 3D] after T
  __bf16* wlat2 = (__bf16*)alloc((size_t)2 * D * D * 2);
  __bf16* wans1 = (__bf16*)alloc((size_t)2 * D * 2 * D * 2);
  __bf16* wans2 = (__bf16*)alloc((size_t)2 * D * D * 2);
  __bf16* wsoul = (__bf16*)alloc((size_t)4 * D * D * 2);
  __bf16* wqb   = (__bf16*)alloc((size_t)D * D * 2);
  __bf16* wkb   = (__bf16*)alloc((size_t)D * D * 2);
  __bf16* wvb   = (__bf16*)alloc((size_t)D * D * 2);
  __bf16* wob   = (__bf16*)alloc((size_t)D * D * 2);
  float*  conf  = (float*)alloc(256);

  auto blocks = [](size_t n) { return dim3((unsigned)((n + 255) / 256)); };

  // ---- setup: weight bf16 conversion + transpose (once; L2-resident) ----
  wt_kernel<<<blocks((size_t)3*D*2*D), 256, 0, stream>>>(lat_W1, wlat1, 3*D, 2*D);
  wt_kernel<<<blocks((size_t)2*D*D),   256, 0, stream>>>(lat_W2, wlat2, 2*D, D);
  wt_kernel<<<blocks((size_t)2*D*2*D), 256, 0, stream>>>(ans_W1, wans1, 2*D, 2*D);
  wt_kernel<<<blocks((size_t)2*D*D),   256, 0, stream>>>(ans_W2, wans2, 2*D, D);
  wt_kernel<<<blocks((size_t)4*D*D),   256, 0, stream>>>(soul_W, wsoul, 4*D, D);
  wt_kernel<<<blocks((size_t)D*D),     256, 0, stream>>>(Wq, wqb, D, D);
  wt_kernel<<<blocks((size_t)D*D),     256, 0, stream>>>(Wk, wkb, D, D);
  wt_kernel<<<blocks((size_t)D*D),     256, 0, stream>>>(Wv, wvb, D, D);
  wt_kernel<<<blocks((size_t)D*D),     256, 0, stream>>>(Wo, wob, D, D);

  copy_kernel<<<blocks((size_t)B*D), 256, 0, stream>>>(y0, outY, (size_t)B*D);
  zero_kernel<<<blocks((size_t)B*D), 256, 0, stream>>>(z, (size_t)B*D);
  zero_kernel<<<1, 64, 0, stream>>>(conf, 8);
  // al depends only on `a` -> compute once
  affect_kernel<<<blocks((size_t)B*D), 256, 0, stream>>>(aIn, aff_W, aff_b, al, (size_t)B*D);

  for (int k = 0; k < KIT; ++k) {
    // ---- 4x recursive latent refinement ----
    for (int it = 0; it < 4; ++it) {
      catln_kernel<3><<<B, 256, 0, stream>>>(x, outY, z, ln_lat_g, ln_lat_b, cat);
      launch_gemm(EPI_GELU_BF16, cat, wlat1, lat_b1, h1, B, 2 * D, 3 * D, 2 * D, stream);
      launch_gemm(EPI_F32,       h1,  wlat2, lat_b2, tmp, B, D, 2 * D, D, stream);
      zresln_kernel<<<B, 256, 0, stream>>>(z, tmp, 0.3f, ln_z_g, ln_z_b);
    }
    // ---- affect gate: z += 0.3*g*al, g = sigmoid([z,al]@gate_W+b) ----
    gate_update_kernel<<<B, 256, 0, stream>>>(z, al, gate_W, gate_b, zbf, conf + k);

    // ---- KV cache for immutable hist entries (exact restructuring) ----
    launch_gemm(EPI_BF16, zbf, wkb, bk,  Kc + (size_t)k * D, B, D, D, 5 * D, stream);
    launch_gemm(EPI_BF16, zbf, wvb, bvb, Vc + (size_t)k * D, B, D, D, 5 * D, stream);
    if (k >= 1) {
      launch_gemm(EPI_BF16, zbf, wqb, bq, qbf, B, D, D, D, stream);
      attn_combine_kernel<<<B, 256, 0, stream>>>(qbf, Kc, Vc, k + 1, obf);
      launch_gemm(EPI_F32, obf, wob, bo, tmp, B, D, D, D, stream);
      zresln_kernel<<<B, 256, 0, stream>>>(z, tmp, 1.0f, ln_z_g, ln_z_b);
    }

    // ---- SoulKernel: m = tanh([x,y,z,r]@soul_W+b); z += 0.3*sigmoid(m@gW+gb)*m ----
    soulcat_kernel<<<blocks((size_t)B*4*D), 256, 0, stream>>>(x, outY, z, cat, (size_t)B*4*D);
    launch_gemm(EPI_TANH_F32, cat, wsoul, soul_b, tmp, B, D, 4 * D, D, stream);
    soulgate_kernel<<<B, 256, 0, stream>>>(z, tmp, soul_gW, soul_gb);

    // ---- answer head: y += 0.4 * MLP(LN([y,z])) ----
    catln_kernel<2><<<B, 256, 0, stream>>>(outY, z, nullptr, ln_ans_g, ln_ans_b, cat);
    launch_gemm(EPI_GELU_BF16, cat, wans1, ans_b1, h1, B, 2 * D, 2 * D, 2 * D, stream);
    launch_gemm(EPI_F32,       h1,  wans2, ans_b2, tmp, B, D, 2 * D, D, stream);
    axpy_kernel<<<blocks((size_t)B*D), 256, 0, stream>>>(outY, tmp, 0.4f, (size_t)B*D);
  }

  // ---- heads ----
  pad_kernel<<<B, 256, 0, stream>>>(z, pad_W, pad_b, outPad);
  conf_final_kernel<<<1, 32, 0, stream>>>(conf, outConf);
}
